// SimpleQuantumLLM_38826504356530
// MI455X (gfx1250) — matrix operand order repaired and checked by reading, not compiled, not run on hardware
//
#include <hip/hip_runtime.h>

typedef __attribute__((ext_vector_type(16))) _Float16 v16h;
typedef __attribute__((ext_vector_type(8)))  _Float16 v8h;
typedef __attribute__((ext_vector_type(8)))  float    v8f;

#define BATCH    4
#define LEN      2048
#define DIM      1024
#define NTHREADS 256

__device__ __forceinline__ float block_sum(float v, float* red) {
  const int t = threadIdx.x;
  red[t] = v;
  __syncthreads();
  for (int s = NTHREADS / 2; s > 0; s >>= 1) {
    if (t < s) red[t] += red[t + s];
    __syncthreads();
  }
  float r = red[0];
  __syncthreads();
  return r;
}

// ---------------------------------------------------------------------------
// Kernel 1: per-row scalar a = 0.01 * D^-1/2 * ( inv*(sum(x*g) - m*sum(g)) + sum(b) )
// for q_real / q_imag / k_real / k_imag  (blockIdx.y selects the array)
// ---------------------------------------------------------------------------
__global__ void __launch_bounds__(NTHREADS)
rowstats_kernel(const float* __restrict__ qr, const float* __restrict__ qi,
                const float* __restrict__ kr, const float* __restrict__ ki,
                const float* __restrict__ qg, const float* __restrict__ qb,
                const float* __restrict__ kg, const float* __restrict__ kb,
                float* __restrict__ out) {
  __shared__ float red[NTHREADS];
  const int which = blockIdx.y;
  const float* x; const float* g; const float* bb;
  if (which == 0)      { x = qr; g = qg; bb = qb; }
  else if (which == 1) { x = qi; g = qg; bb = qb; }
  else if (which == 2) { x = kr; g = kg; bb = kb; }
  else                 { x = ki; g = kg; bb = kb; }
  const size_t row = blockIdx.x;
  const float* xr = x + row * DIM;
  float s1 = 0.f, s2 = 0.f, sgx = 0.f, sg = 0.f, sb = 0.f;
  for (int d = threadIdx.x; d < DIM; d += NTHREADS) {
    float xv = xr[d], gv = g[d], bv = bb[d];
    s1 += xv; s2 += xv * xv; sgx += xv * gv; sg += gv; sb += bv;
  }
  s1  = block_sum(s1, red);
  s2  = block_sum(s2, red);
  sgx = block_sum(sgx, red);
  sg  = block_sum(sg, red);
  sb  = block_sum(sb, red);
  if (threadIdx.x == 0) {
    float m   = s1 * (1.0f / DIM);
    float inv = rsqrtf(s2 * (1.0f / DIM) - m * m + 1e-5f);
    const float sc = 0.01f * 0.03125f;  // 0.01 * DIM^{-1/2}, DIM = 1024
    out[(size_t)which * (BATCH * LEN) + row] = sc * (inv * (sgx - m * sg) + sb);
  }
}

// ---------------------------------------------------------------------------
// Kernel 2: LN(v)*0.01 -> f16, row-major [B,L,D]; blockIdx.y: 0=real 1=imag
// ---------------------------------------------------------------------------
__global__ void __launch_bounds__(NTHREADS)
vnorm_kernel(const float* __restrict__ vr, const float* __restrict__ vi,
             const float* __restrict__ g, const float* __restrict__ b,
             _Float16* __restrict__ vrh, _Float16* __restrict__ vih) {
  __shared__ float red[NTHREADS];
  const size_t row = blockIdx.x;
  const float* x = (blockIdx.y ? vi : vr) + row * DIM;
  _Float16*    o = (blockIdx.y ? vih : vrh) + row * DIM;
  float vals[DIM / NTHREADS];
  float s1 = 0.f, s2 = 0.f;
#pragma unroll
  for (int u = 0; u < DIM / NTHREADS; ++u) {
    float v = x[threadIdx.x + u * NTHREADS];
    vals[u] = v; s1 += v; s2 += v * v;
  }
  s1 = block_sum(s1, red);
  s2 = block_sum(s2, red);
  float m   = s1 * (1.0f / DIM);
  float inv = rsqrtf(s2 * (1.0f / DIM) - m * m + 1e-5f);
#pragma unroll
  for (int u = 0; u < DIM / NTHREADS; ++u) {
    int d = threadIdx.x + u * NTHREADS;
    o[d] = (_Float16)(((vals[u] - m) * inv * g[d] + b[d]) * 0.01f);
  }
}

// ---------------------------------------------------------------------------
// Kernel 3: one block per (batch, query row): scores s = 10*x*rsqrt(x^2+y^2),
// bounded |s|<=10 so softmax needs no max pass. Write normalized row as f16.
// ---------------------------------------------------------------------------
__global__ void __launch_bounds__(NTHREADS)
attn_kernel(const float* __restrict__ rowsums, _Float16* __restrict__ Ah) {
  __shared__ float red[NTHREADS];
  const int b = blockIdx.y;
  const int i = blockIdx.x;
  const float* aqr = rowsums;
  const float* aqi = rowsums + (size_t)BATCH * LEN;
  const float* akr = rowsums + 2 * (size_t)BATCH * LEN;
  const float* aki = rowsums + 3 * (size_t)BATCH * LEN;
  const float xr0 = aqr[(size_t)b * LEN + i] + 1e-8f;
  const float xi0 = aqi[(size_t)b * LEN + i] + 1e-8f;
  float e[LEN / NTHREADS];
  float acc = 0.f;
#pragma unroll
  for (int u = 0; u < LEN / NTHREADS; ++u) {
    int j = threadIdx.x + u * NTHREADS;
    float xx = xr0 - akr[(size_t)b * LEN + j];
    float yy = xi0 - aki[(size_t)b * LEN + j];
    float s  = 10.f * xx * rsqrtf(xx * xx + yy * yy);  // 10*cos(atan2(yy,xx))
    float ev = __expf(s);
    e[u] = ev; acc += ev;
  }
  acc = block_sum(acc, red);
  float zinv = 1.0f / acc;
  _Float16* arow = Ah + ((size_t)b * LEN + i) * LEN;
#pragma unroll
  for (int u = 0; u < LEN / NTHREADS; ++u)
    arow[threadIdx.x + u * NTHREADS] = (_Float16)(e[u] * zinv);
}

// ---------------------------------------------------------------------------
// Kernel 4: O = 0.1 * A @ V  (real and imag share the A operand).
// Block = 8 waves; tile M=128 (8 WMMA M-sub-tiles per wave), N=128
// (16 cols/wave), K-step 32.  v_wmma_f32_16x16x32_f16, f32 accumulation.
// Halves L2 V-traffic vs an M=64 tile: 512 blocks x (512KB A + 1MB V) ~ 0.77GB.
// A-operand lane layout (16-bit A 16x32): lanes 0-15 row M=lane, K 0-7 & 16-23;
// lanes 16-31 row M=lane-16, K 8-15 & 24-31 -> two 16B loads per sub-tile.
// B-operand: lane l holds V col n0+(l&15), K same as A (8h+0..7, 16+8h+0..7).
// ---------------------------------------------------------------------------
__global__ void __launch_bounds__(NTHREADS)
gemm_kernel(const _Float16* __restrict__ Ah, const _Float16* __restrict__ Vrh,
            const _Float16* __restrict__ Vih, float* __restrict__ Or,
            float* __restrict__ Oi) {
  const int b    = blockIdx.z;
  const int m0   = blockIdx.y * 128;
  const int wave = threadIdx.x >> 5;
  const int lane = threadIdx.x & 31;
  const int n0   = blockIdx.x * 128 + wave * 16;

  const _Float16* Vr = Vrh + (size_t)b * LEN * DIM + n0;
  const _Float16* Vi = Vih + (size_t)b * LEN * DIM + n0;

  const int am = lane & 15;         // A row within 16-row sub-tile
  const int ak = (lane >> 4) << 3;  // K phase: 0 or 8
  const _Float16* ap0 =
      Ah + ((size_t)b * LEN + m0 + am) * LEN + ak;  // per-lane A base

  v8f cr[8] = {};
  v8f ci[8] = {};

  for (int k0 = 0; k0 < LEN; k0 += 32) {
    v16h bvr, bvi;
#pragma unroll
    for (int s = 0; s < 16; ++s) {
      const size_t kk = (size_t)(k0 + ak + (s & 7) + ((s >> 3) << 4));
      bvr[s] = Vr[kk * DIM + am];
      bvi[s] = Vi[kk * DIM + am];
    }
#pragma unroll
    for (int mi = 0; mi < 8; ++mi) {
      const _Float16* ap = ap0 + (size_t)(mi * 16) * LEN + k0;
      v8h alo = *(const v8h*)(ap);
      v8h ahi = *(const v8h*)(ap + 16);
      v16h a;
#pragma unroll
      for (int e = 0; e < 8; ++e) { a[e] = alo[e]; a[e + 8] = ahi[e]; }
      cr[mi] = __builtin_amdgcn_wmma_f32_16x16x32_f16(
          false, a, false, bvr, (short)0, cr[mi], false, false);
      ci[mi] = __builtin_amdgcn_wmma_f32_16x16x32_f16(
          false, a, false, bvi, (short)0, ci[mi], false, false);
    }
  }

  // C/D layout: VGPR r -> (M = r + (lane>=16 ? 8 : 0), N = lane&15)
  const int rbase = m0 + ((lane >> 4) << 3);
  const int ocol  = n0 + (lane & 15);
#pragma unroll
  for (int mi = 0; mi < 8; ++mi) {
    float* orp = Or + ((size_t)b * LEN + rbase + mi * 16) * DIM + ocol;
    float* oip = Oi + ((size_t)b * LEN + rbase + mi * 16) * DIM + ocol;
#pragma unroll
    for (int r = 0; r < 8; ++r) {
      orp[(size_t)r * DIM] = cr[mi][r] * 0.1f;
      oip[(size_t)r * DIM] = ci[mi][r] * 0.1f;
    }
  }
}

// ---------------------------------------------------------------------------
// Kernel 5: output LayerNorm into d_out (real block then imag block).
// ---------------------------------------------------------------------------
__global__ void __launch_bounds__(NTHREADS)
outln_kernel(const float* __restrict__ Or, const float* __restrict__ Oi,
             const float* __restrict__ g, const float* __restrict__ b,
             float* __restrict__ out) {
  __shared__ float red[NTHREADS];
  const size_t row = blockIdx.x;
  const size_t BLD = (size_t)BATCH * LEN * DIM;
  const float* x = (blockIdx.y ? Oi : Or) + row * DIM;
  float*       o = out + (blockIdx.y ? BLD : 0) + row * DIM;
  float vals[DIM / NTHREADS];
  float s1 = 0.f, s2 = 0.f;
#pragma unroll
  for (int u = 0; u < DIM / NTHREADS; ++u) {
    float v = x[threadIdx.x + u * NTHREADS];
    vals[u] = v; s1 += v; s2 += v * v;
  }
  s1 = block_sum(s1, red);
  s2 = block_sum(s2, red);
  float m   = s1 * (1.0f / DIM);
  float inv = rsqrtf(s2 * (1.0f / DIM) - m * m + 1e-5f);
#pragma unroll
  for (int u = 0; u < DIM / NTHREADS; ++u) {
    int d = threadIdx.x + u * NTHREADS;
    o[d] = (vals[u] - m) * inv * g[d] + b[d];
  }
}

// ---------------------------------------------------------------------------
extern "C" void kernel_launch(void* const* d_in, const int* in_sizes, int n_in,
                              void* d_out, int out_size, void* d_ws, size_t ws_size,
                              hipStream_t stream) {
  const float* q_real = (const float*)d_in[0];
  const float* q_imag = (const float*)d_in[1];
  const float* k_real = (const float*)d_in[2];
  const float* k_imag = (const float*)d_in[3];
  const float* v_real = (const float*)d_in[4];
  const float* v_imag = (const float*)d_in[5];
  const float* qn_g = (const float*)d_in[6];
  const float* qn_b = (const float*)d_in[7];
  const float* kn_g = (const float*)d_in[8];
  const float* kn_b = (const float*)d_in[9];
  const float* vn_g = (const float*)d_in[10];
  const float* vn_b = (const float*)d_in[11];
  const float* on_g = (const float*)d_in[12];
  const float* on_b = (const float*)d_in[13];

  char* ws = (char*)d_ws;
  const size_t BL  = (size_t)BATCH * LEN;
  const size_t BLD = BL * DIM;
  size_t off = 0;
  float*    rowsums = (float*)(ws + off);    off += 4 * BL * sizeof(float); // 128 KB
  _Float16* Vrh     = (_Float16*)(ws + off); off += BLD * sizeof(_Float16); // 16 MB
  _Float16* Vih     = (_Float16*)(ws + off); off += BLD * sizeof(_Float16); // 16 MB
  _Float16* Ah      = (_Float16*)(ws + off); off += BL * LEN * sizeof(_Float16); // 32 MB
  float*    Or      = (float*)(ws + off);    off += BLD * sizeof(float);    // 32 MB
  float*    Oi      = (float*)(ws + off);    off += BLD * sizeof(float);    // 32 MB
  (void)off; (void)ws_size; (void)in_sizes; (void)n_in; (void)out_size;

  rowstats_kernel<<<dim3((unsigned)BL, 4), NTHREADS, 0, stream>>>(
      q_real, q_imag, k_real, k_imag, qn_g, qn_b, kn_g, kn_b, rowsums);
  vnorm_kernel<<<dim3((unsigned)BL, 2), NTHREADS, 0, stream>>>(
      v_real, v_imag, vn_g, vn_b, Vrh, Vih);
  attn_kernel<<<dim3(LEN, BATCH), NTHREADS, 0, stream>>>(rowsums, Ah);
  gemm_kernel<<<dim3(DIM / 128, LEN / 128, BATCH), NTHREADS, 0, stream>>>(
      Ah, Vrh, Vih, Or, Oi);
  outln_kernel<<<dim3((unsigned)BL, 2), NTHREADS, 0, stream>>>(
      Or, Oi, on_g, on_b, (float*)d_out);
}
